// LSM_8409545965998
// MI455X (gfx1250) — matrix operand order repaired and checked by reading, not compiled, not run on hardware
//
#include <hip/hip_runtime.h>

typedef __attribute__((ext_vector_type(8)))  _Float16 v8h;
typedef __attribute__((ext_vector_type(16))) _Float16 v16h;
typedef __attribute__((ext_vector_type(8)))  float    v8f;

#define N_NEU   2000
#define IN_SZ   2312
#define KPAD1   2336   // 73*32
#define KPAD2   2016   // 63*32
#define BATCH   128
#define TSTEPS  100
#define M1      (BATCH*TSTEPS)   // 12800

static __device__ __forceinline__ v16h cat8(v8h lo, v8h hi) {
    return __builtin_shufflevector(lo, hi, 0,1,2,3,4,5,6,7,8,9,10,11,12,13,14,15);
}
static __device__ __forceinline__ v8f wmma16(v16h a, v16h b, v8f c) {
    // D = A(16x32,f16) * B(32x16,f16) + C(16x16,f32)
    return __builtin_amdgcn_wmma_f32_16x16x32_f16(false, a, false, b, (short)0, c, false, false);
}

// CDNA5 async copy: global -> LDS, 16B per lane, tracked by ASYNCcnt (no VGPR round trip)
static __device__ __forceinline__ void async_g2l_b128(unsigned lds_off, const void* gaddr) {
    asm volatile("global_load_async_to_lds_b128 %0, %1, off"
                 :: "v"(lds_off), "v"(gaddr) : "memory");
}
static __device__ __forceinline__ void wait_async_le4() {
    asm volatile("s_wait_asynccnt 0x4" ::: "memory");
}
static __device__ __forceinline__ void wait_async_0() {
    asm volatile("s_wait_asynccnt 0x0" ::: "memory");
}
#define LDS_OFF(p) ((unsigned)(unsigned long long)(const void*)(p))

// ---------------- convert f32 -> f16 with K padding (zeros) ----------------
__global__ void cvt_pad_f16(const float* __restrict__ src, _Float16* __restrict__ dst,
                            int rows, int scols, int dcols) {
    long total = (long)rows * dcols;
    for (long i = (long)blockIdx.x * blockDim.x + threadIdx.x; i < total;
         i += (long)gridDim.x * blockDim.x) {
        long r = i / dcols;
        int  c = (int)(i - r * dcols);
        dst[i] = (c < scols) ? (_Float16)src[r * scols + c] : (_Float16)0.0f;
    }
}

__global__ void zero_u32(unsigned* __restrict__ p, long n) {
    for (long i = (long)blockIdx.x * blockDim.x + threadIdx.x; i < n;
         i += (long)gridDim.x * blockDim.x) p[i] = 0u;
}

// ---------------- phase 1: curr = xf @ Win^T + b_in ----------------
// Block tile 128x128, 8 waves as 2(M) x 4(N), each wave 64x32 = 4x2 WMMA frags.
// Double-buffered LDS tiles filled by GLOBAL_LOAD_ASYNC_TO_LDS_B128 (ASYNCcnt).
#define BM 128
#define BN 128
#define BK 32
#define LDSS 40   // LDS row stride in elements (80B, 16B aligned, bank-skewed)
#define NT1 (KPAD1 / BK)   // 73 K-tiles

__launch_bounds__(256)
__global__ void gemm_curr(const _Float16* __restrict__ A,   // [M1][KPAD1]
                          const _Float16* __restrict__ Bt,  // Win f16 [N_NEU][KPAD1]
                          const float*    __restrict__ bias,
                          float*          __restrict__ C)   // [M1][N_NEU]
{
    __shared__ __align__(16) _Float16 As[2][BM * LDSS];
    __shared__ __align__(16) _Float16 Bs[2][BN * LDSS];

    const int tid  = threadIdx.x;
    const int lane = tid & 31;
    const int wid  = tid >> 5;
    const int wm   = wid >> 2;          // 0..1
    const int wn   = wid & 3;           // 0..3
    const int m0   = blockIdx.y * BM;
    const int n0   = blockIdx.x * BN;

    const int srow = tid >> 1;          // staging row 0..127
    const int sseg = tid & 1;           // 16-element half

    // Per-thread staging addresses (32B of A tile + 32B of B tile per K-step).
    // B edge: clamp the Win row; garbage lands only in output columns >= N_NEU,
    // which the epilogue discards -> inner loop stays branch-free, EXEC all-ones.
    int gn = n0 + srow; if (gn >= N_NEU) gn = N_NEU - 1;
    const _Float16* gA = A  + (size_t)(m0 + srow) * KPAD1 + sseg * 16;
    const _Float16* gB = Bt + (size_t)gn          * KPAD1 + sseg * 16;
    const unsigned  lA0 = LDS_OFF(&As[0][srow * LDSS + sseg * 16]);
    const unsigned  lB0 = LDS_OFF(&Bs[0][srow * LDSS + sseg * 16]);
    const unsigned  lA1 = LDS_OFF(&As[1][srow * LDSS + sseg * 16]);
    const unsigned  lB1 = LDS_OFF(&Bs[1][srow * LDSS + sseg * 16]);

    v8f c[4][2] = {};

    const int ar = lane & 15;
    const int ka = (lane < 16) ? 0 : 8;    // A-frag K base (ISA 16-bit A layout)
    const int kb = (lane < 16) ? 0 : 8;    // B-frag K base (same K order as A)

    // prologue: async-fill tile 0 into buffer 0 (4 x b128 per thread)
    async_g2l_b128(lA0,      gA);
    async_g2l_b128(lA0 + 16, gA + 8);
    async_g2l_b128(lB0,      gB);
    async_g2l_b128(lB0 + 16, gB + 8);

    for (int it = 0; it < NT1; ++it) {
        const int buf = it & 1;
        if (it + 1 < NT1) {
            // overlap: issue next tile's async copies into the other buffer
            const _Float16* nA = gA + (size_t)(it + 1) * BK;
            const _Float16* nB = gB + (size_t)(it + 1) * BK;
            const unsigned  nlA = buf ? lA0 : lA1;
            const unsigned  nlB = buf ? lB0 : lB1;
            async_g2l_b128(nlA,      nA);
            async_g2l_b128(nlA + 16, nA + 8);
            async_g2l_b128(nlB,      nB);
            async_g2l_b128(nlB + 16, nB + 8);
            wait_async_le4();   // in-order completion: current tile's 4 loads landed
        } else {
            wait_async_0();
        }
        __syncthreads();        // all waves' async fills of this buffer visible

        v16h a[4], b[2];
        #pragma unroll
        for (int i = 0; i < 4; ++i) {
            const _Float16* p = &As[buf][(wm * 64 + i * 16 + ar) * LDSS + ka];
            a[i] = cat8(*(const v8h*)p, *(const v8h*)(p + 16));  // K {k..k+7, k+16..k+23}
        }
        #pragma unroll
        for (int j = 0; j < 2; ++j) {
            const _Float16* p = &Bs[buf][(wn * 32 + j * 16 + ar) * LDSS + kb];
            b[j] = cat8(*(const v8h*)p, *(const v8h*)(p + 16));  // K {k..k+7, k+16..k+23}
        }
        #pragma unroll
        for (int i = 0; i < 4; ++i)
            #pragma unroll
            for (int j = 0; j < 2; ++j)
                c[i][j] = wmma16(a[i], b[j], c[i][j]);
        __syncthreads();        // done reading this buffer before it is refilled
    }

    // ---- epilogue: C[m][n] = acc + b_in[n] ----
    const int mlane = (lane < 16) ? 0 : 8;   // f32 C/D layout: vgpr r -> M = r (+8 hi lanes)
    const int ncl   = lane & 15;
    #pragma unroll
    for (int i = 0; i < 4; ++i)
        #pragma unroll
        for (int j = 0; j < 2; ++j) {
            int ng = n0 + wn * 32 + j * 16 + ncl;
            if (ng >= N_NEU) continue;
            float bv = bias[ng];
            int mbase = m0 + wm * 64 + i * 16 + mlane;
            #pragma unroll
            for (int r = 0; r < 8; ++r)
                C[(size_t)(mbase + r) * N_NEU + ng] = c[i][j][r] + bv;
        }
}

// ---------------- phase 2: one recurrent step (fused GEMM + RSynaptic) ----------------
// Each wave computes one 16x16 tile of rc = spk_prev @ Wlsm^T, then updates syn/mem/spk.
// Operands are L2-resident (spk 0.5MB, Wlsm f16 8MB << 192MB L2): direct global frags.
__launch_bounds__(128)
__global__ void rec_step(const _Float16* __restrict__ spk_in,  // [BATCH][KPAD2]
                         _Float16*       __restrict__ spk_out, // [BATCH][KPAD2]
                         const _Float16* __restrict__ W,       // Wlsm f16 [N_NEU][KPAD2]
                         const float*    __restrict__ curr,    // [M1][N_NEU], row = b*T + t
                         const float*    __restrict__ brec,
                         float*          __restrict__ syn,     // [BATCH][N_NEU]
                         float*          __restrict__ mem,     // [BATCH][N_NEU]
                         float*          __restrict__ out,     // [T][BATCH][N_NEU]
                         int t)
{
    const int lane = threadIdx.x & 31;
    const int wid  = threadIdx.x >> 5;
    const int tile = blockIdx.x * 4 + wid;        // 0..999
    const int mt = tile / 125;                    // 8 M tiles
    const int nt = tile - mt * 125;               // 125 N tiles

    const int ar = lane & 15;
    const int ka = (lane < 16) ? 0 : 8;
    const int kb = (lane < 16) ? 0 : 8;
    const _Float16* arow = spk_in + (size_t)(mt * 16 + ar) * KPAD2;
    const _Float16* brow = W      + (size_t)(nt * 16 + ar) * KPAD2;

    v8f c = {};
    #pragma unroll 4
    for (int kk = 0; kk < KPAD2; kk += 32) {
        const _Float16* pa = arow + kk + ka;
        const _Float16* pb = brow + kk + kb;
        v16h a = cat8(*(const v8h*)pa, *(const v8h*)(pa + 16));
        v16h b = cat8(*(const v8h*)pb, *(const v8h*)(pb + 16));
        c = wmma16(a, b, c);
    }

    const int mbase = mt * 16 + ((lane < 16) ? 0 : 8);
    const int ng    = nt * 16 + (lane & 15);
    const float bv  = brec[ng];
    #pragma unroll
    for (int r = 0; r < 8; ++r) {
        int    mb   = mbase + r;                            // batch index
        size_t sidx = (size_t)mb * N_NEU + ng;
        float  cur  = curr[((size_t)mb * TSTEPS + t) * N_NEU + ng];
        float  so   = syn[sidx];
        float  mo   = mem[sidx];
        float  reset = (mo - 1.0f > 0.0f) ? 1.0f : 0.0f;    // snntorch: reset from prev mem
        float  sn   = 0.9f * so + cur + c[r] + bv;          // ALPHA
        float  mn   = 0.8f * mo + sn - reset;               // BETA, TH=1
        float  s    = (mn - 1.0f > 0.0f) ? 1.0f : 0.0f;     // spike
        syn[sidx] = sn;
        mem[sidx] = mn;
        out[((size_t)t * BATCH + mb) * N_NEU + ng] = s;
        spk_out[(size_t)mb * KPAD2 + ng] = (_Float16)s;
    }
}

// ---------------- host ----------------
extern "C" void kernel_launch(void* const* d_in, const int* in_sizes, int n_in,
                              void* d_out, int out_size, void* d_ws, size_t ws_size,
                              hipStream_t stream) {
    (void)in_sizes; (void)n_in; (void)out_size; (void)ws_size;
    const float* x     = (const float*)d_in[0];
    const float* Win   = (const float*)d_in[1];
    const float* b_in  = (const float*)d_in[2];
    const float* Wlsm  = (const float*)d_in[3];
    const float* b_rec = (const float*)d_in[4];
    float* out = (float*)d_out;

    char* ws = (char*)d_ws;
    size_t off = 0;
    auto carve = [&](size_t bytes) -> char* {
        char* p = ws + off;
        off = (off + bytes + 255) & ~(size_t)255;
        return p;
    };
    _Float16* xh   = (_Float16*)carve((size_t)M1    * KPAD1 * 2);
    _Float16* winh = (_Float16*)carve((size_t)N_NEU * KPAD1 * 2);
    _Float16* wlh  = (_Float16*)carve((size_t)N_NEU * KPAD2 * 2);
    float*    curr = (float*)   carve((size_t)M1    * N_NEU * 4);
    char*     zbeg = ws + off;                                   // state region start
    float*    syn  = (float*)   carve((size_t)BATCH * N_NEU * 4);
    float*    mem  = (float*)   carve((size_t)BATCH * N_NEU * 4);
    _Float16* spk0 = (_Float16*)carve((size_t)BATCH * KPAD2 * 2);
    _Float16* spk1 = (_Float16*)carve((size_t)BATCH * KPAD2 * 2);
    size_t zbytes = (size_t)((ws + off) - zbeg);

    // f32 -> f16 (K-padded with zeros)
    cvt_pad_f16<<<2048, 256, 0, stream>>>(x,    xh,   M1,    IN_SZ, KPAD1);
    cvt_pad_f16<<<1024, 256, 0, stream>>>(Win,  winh, N_NEU, IN_SZ, KPAD1);
    cvt_pad_f16<<<1024, 256, 0, stream>>>(Wlsm, wlh,  N_NEU, N_NEU, KPAD2);
    // zero syn/mem/spk ping-pong buffers (fresh state each launch)
    zero_u32<<<1024, 256, 0, stream>>>((unsigned*)zbeg, (long)(zbytes / 4));

    // phase 1: big GEMM (compute-bound, f16 WMMA, f32 accumulate, async-LDS pipeline)
    dim3 g1((N_NEU + BN - 1) / BN, M1 / BM);   // 16 x 100
    gemm_curr<<<g1, 256, 0, stream>>>(xh, winh, b_in, curr);

    // phase 2: 100 sequential fused recurrent steps (spike buffers ping-pong)
    for (int t = 0; t < TSTEPS; ++t) {
        const _Float16* si = (t & 1) ? spk1 : spk0;
        _Float16*       so = (t & 1) ? spk0 : spk1;
        rec_step<<<250, 128, 0, stream>>>(si, so, wlh, curr, b_rec, syn, mem, out, t);
    }
}
